// TransformerBlock_8641474200325
// MI455X (gfx1250) — matrix operand order repaired and checked by reading, not compiled, not run on hardware
//
#include <hip/hip_runtime.h>

// ---------------------------------------------------------------------------
// Transformer block for MI455X (gfx1250, wave32, WMMA 16x16x32 f16 -> f32 acc)
// B=2, N=2048, D=768, H=12, HD=64, EXP=4
// ---------------------------------------------------------------------------

typedef __attribute__((ext_vector_type(16))) _Float16 v16h;
typedef __attribute__((ext_vector_type(8)))  _Float16 v8h;
typedef __attribute__((ext_vector_type(8)))  float    v8f;

#define BDIM 768
#define NSEQ 2048
#define NHEAD 12
#define HDIM 64
#define ROWS 4096          // B*N
#define QKVCOLS 2304       // 3*D
#define FFCOLS 3072        // EXP*D

// CDNA5 async global->LDS copy path (toolchain-dependent availability)
#if defined(__has_builtin)
#if __has_builtin(__builtin_amdgcn_global_load_async_to_lds_b128) && \
    __has_builtin(__builtin_amdgcn_s_wait_asynccnt)
#define USE_ASYNC_LDS 1
#endif
#endif

#ifdef USE_ASYNC_LDS
// b128 payload type exactly as the builtin expects: GCC-style int 4-vector.
typedef int v4i_g __attribute__((__vector_size__(4 * sizeof(int))));
typedef __attribute__((address_space(1))) v4i_g as1_v4i;
typedef __attribute__((address_space(3))) v4i_g as3_v4i;
static __device__ __forceinline__ void async_b128(const void* g, void* l) {
  __builtin_amdgcn_global_load_async_to_lds_b128(
      (as1_v4i*)g, (as3_v4i*)l, 0, 0);
}
#endif

static __device__ __forceinline__ v16h mk16(v8h lo, v8h hi) {
  v16h r;
#pragma unroll
  for (int i = 0; i < 8; ++i) { r[i] = lo[i]; r[i + 8] = hi[i]; }
  return r;
}

static __device__ __forceinline__ v8f wmma16x16x32(v16h a, v16h b, v8f c) {
  return __builtin_amdgcn_wmma_f32_16x16x32_f16(
      false, a, false, b, (short)0, c, false, false);
}

static __device__ __forceinline__ float gelu_exact(float v) {
  return 0.5f * v * (1.0f + erff(v * 0.70710678118654752f));
}

// ---------------------------------------------------------------------------
__global__ void cvt_f32_to_f16(const float* __restrict__ s,
                               _Float16* __restrict__ d, int n) {
  int i = blockIdx.x * 256 + threadIdx.x;
  if (i < n) d[i] = (_Float16)s[i];
}

// ---------------------------------------------------------------------------
// LayerNorm rows of 768 -> f16. One 256-thread block per row.
// ---------------------------------------------------------------------------
__global__ __launch_bounds__(256)
void ln_kernel(const float* __restrict__ x, const float* __restrict__ g,
               const float* __restrict__ bb, _Float16* __restrict__ out) {
  __shared__ float red[256];
  const int row = blockIdx.x;
  const int t = threadIdx.x;
  const float* xr = x + (size_t)row * BDIM;

  float s = 0.f;
  for (int i = t; i < BDIM; i += 256) s += xr[i];
  red[t] = s; __syncthreads();
  for (int off = 128; off > 0; off >>= 1) {
    if (t < off) red[t] += red[t + off];
    __syncthreads();
  }
  const float mu = red[0] * (1.0f / BDIM);
  __syncthreads();

  float v = 0.f;
  for (int i = t; i < BDIM; i += 256) { float d = xr[i] - mu; v += d * d; }
  red[t] = v; __syncthreads();
  for (int off = 128; off > 0; off >>= 1) {
    if (t < off) red[t] += red[t + off];
    __syncthreads();
  }
  const float rstd = rsqrtf(red[0] * (1.0f / BDIM) + 1e-5f);

  for (int i = t; i < BDIM; i += 256)
    out[(size_t)row * BDIM + i] = (_Float16)((xr[i] - mu) * rstd * g[i] + bb[i]);
}

// ---------------------------------------------------------------------------
// WMMA GEMM: C[M,N] = A16[M,K] @ B16[K,N] + bias (+resid) (+GELU)
// 256 threads = 8 waves; 128x128 block tile; K staged 32-deep, double-buffered
// LDS with software pipelining (async global->LDS for A when available).
// outMode 0: write outF/outH [M,N].
// outMode 2: deinterleave QKV epilogue into q[h][row][d], k[h][row][d],
//            vT[h][d][row]  (col = h*192 + d*3 + m, m in {q,k,v}).
// ---------------------------------------------------------------------------
__global__ __launch_bounds__(256)
void gemm_wmma(const _Float16* __restrict__ A, const _Float16* __restrict__ Bm,
               const float* __restrict__ bias, const float* __restrict__ resid,
               float* __restrict__ outF, _Float16* __restrict__ outH,
               _Float16* __restrict__ qOut, _Float16* __restrict__ kOut,
               _Float16* __restrict__ vTOut,
               int M, int N, int K, int act, int outMode) {
  __shared__ alignas(32) _Float16 As[2][128][32];   // [m][k]
  __shared__ alignas(32) _Float16 BsT[2][128][32];  // [n][k] transposed

  const int t = threadIdx.x;
  const int lane = t & 31, w = t >> 5;
  const int r = lane & 15, half = lane >> 4;
  const int mBase = blockIdx.y * 128;
  const int nBase = blockIdx.x * 128;

  // fixed per-thread staging coordinates (2 chunks each for A and B)
  const int aRow = t >> 2, aC8 = (t & 3) * 8;       // + rows {0,64}
  const int bK = t >> 4, bN8 = (t & 15) * 8;        // + ks   {0,16}

  v8f acc[8];
#pragma unroll
  for (int i = 0; i < 8; ++i) acc[i] = (v8f){0,0,0,0,0,0,0,0};

#ifndef USE_ASYNC_LDS
  v8h aR0, aR1;
#endif
  v8h bR0, bR1;

  // ---- prologue fetch (kt = 0, buf = 0) ----
  {
    const int kt = 0;
#ifdef USE_ASYNC_LDS
    async_b128(&A[(size_t)(mBase + aRow) * K + kt + aC8], &As[0][aRow][aC8]);
    async_b128(&A[(size_t)(mBase + aRow + 64) * K + kt + aC8], &As[0][aRow + 64][aC8]);
#else
    aR0 = *(const v8h*)&A[(size_t)(mBase + aRow) * K + kt + aC8];
    aR1 = *(const v8h*)&A[(size_t)(mBase + aRow + 64) * K + kt + aC8];
#endif
    bR0 = *(const v8h*)&Bm[(size_t)(kt + bK) * N + nBase + bN8];
    bR1 = *(const v8h*)&Bm[(size_t)(kt + bK + 16) * N + nBase + bN8];
#ifdef USE_ASYNC_LDS
    __builtin_amdgcn_s_wait_asynccnt(0);
#else
    *(v8h*)&As[0][aRow][aC8] = aR0;
    *(v8h*)&As[0][aRow + 64][aC8] = aR1;
#endif
#pragma unroll
    for (int j = 0; j < 8; ++j) BsT[0][bN8 + j][bK] = bR0[j];
#pragma unroll
    for (int j = 0; j < 8; ++j) BsT[0][bN8 + j][bK + 16] = bR1[j];
  }
  __syncthreads();

  int buf = 0;
  for (int kt = 0; kt < K; kt += 32, buf ^= 1) {
    const bool hasNext = (kt + 32) < K;
    if (hasNext) {
      const int kn = kt + 32;
#ifdef USE_ASYNC_LDS
      async_b128(&A[(size_t)(mBase + aRow) * K + kn + aC8], &As[buf ^ 1][aRow][aC8]);
      async_b128(&A[(size_t)(mBase + aRow + 64) * K + kn + aC8], &As[buf ^ 1][aRow + 64][aC8]);
#else
      aR0 = *(const v8h*)&A[(size_t)(mBase + aRow) * K + kn + aC8];
      aR1 = *(const v8h*)&A[(size_t)(mBase + aRow + 64) * K + kn + aC8];
#endif
      bR0 = *(const v8h*)&Bm[(size_t)(kn + bK) * N + nBase + bN8];
      bR1 = *(const v8h*)&Bm[(size_t)(kn + bK + 16) * N + nBase + bN8];
      if (kn + 32 < K) {  // hint tile-after-next toward L2
        __builtin_prefetch(&A[(size_t)(mBase + aRow) * K + kn + 32 + aC8], 0, 0);
        __builtin_prefetch(&Bm[(size_t)(kn + 32 + bK) * N + nBase + bN8], 0, 0);
      }
    }

    // ---- compute on buffer `buf` ----
    const _Float16* ap = &As[buf][w * 16 + r][0];
    v16h afr = *(const v16h*)(ap + 16 * half);
    v16h bfr[8];
#pragma unroll
    for (int nt = 0; nt < 8; ++nt)
      bfr[nt] = *(const v16h*)&BsT[buf][nt * 16 + r][16 * half];
#pragma unroll
    for (int nt = 0; nt < 8; ++nt)
      acc[nt] = wmma16x16x32(afr, bfr[nt], acc[nt]);

    if (hasNext) {
#ifdef USE_ASYNC_LDS
      __builtin_amdgcn_s_wait_asynccnt(0);
#else
      *(v8h*)&As[buf ^ 1][aRow][aC8] = aR0;
      *(v8h*)&As[buf ^ 1][aRow + 64][aC8] = aR1;
#endif
#pragma unroll
      for (int j = 0; j < 8; ++j) BsT[buf ^ 1][bN8 + j][bK] = bR0[j];
#pragma unroll
      for (int j = 0; j < 8; ++j) BsT[buf ^ 1][bN8 + j][bK + 16] = bR1[j];
    }
    __syncthreads();
  }

  // ---- epilogue: C layout vgpr i -> row i + 8*half, col = lane%16 ----
  const int mRow0 = mBase + w * 16 + half * 8;
  if (outMode == 2) {
#pragma unroll
    for (int nt = 0; nt < 8; ++nt) {
      const int col = nBase + nt * 16 + r;
      const int hh = col / 192;
      const int rem = col - hh * 192;
      const int d = rem / 3;
      const int m = rem - d * 3;
      const float bval = bias[col];
#pragma unroll
      for (int i = 0; i < 8; ++i) {
        const int row = mRow0 + i;
        const _Float16 v = (_Float16)(acc[nt][i] + bval);
        if (m == 0)      qOut[((size_t)hh * ROWS + row) * HDIM + d] = v;
        else if (m == 1) kOut[((size_t)hh * ROWS + row) * HDIM + d] = v;
        else             vTOut[((size_t)hh * HDIM + d) * ROWS + row] = v;
      }
    }
  } else {
#pragma unroll
    for (int nt = 0; nt < 8; ++nt) {
      const int col = nBase + nt * 16 + r;
      const float bval = bias[col];
#pragma unroll
      for (int i = 0; i < 8; ++i) {
        const int row = mRow0 + i;
        float v = acc[nt][i] + bval;
        if (act) v = gelu_exact(v);
        if (resid) v += resid[(size_t)row * N + col];
        if (outF) outF[(size_t)row * N + col] = v;
        if (outH) outH[(size_t)row * N + col] = (_Float16)v;
      }
    }
  }
}

// ---------------------------------------------------------------------------
// Flash attention. Grid: (N/128, H, B). Block 256 = 8 waves.
// q16/k16: [h][row][d] contiguous; vT16: [h][d][row] (pre-transposed).
// Each wave owns a 16-row query strip; streams 64-key blocks via LDS.
// Output attn16[4096][768], col = h*64 + d.
// ---------------------------------------------------------------------------
__global__ __launch_bounds__(256)
void attn_flash(const _Float16* __restrict__ q16, const _Float16* __restrict__ k16,
                const _Float16* __restrict__ vT16, _Float16* __restrict__ attn16) {
  __shared__ alignas(32) _Float16 Qs[128][64];      // [q][d]
  __shared__ alignas(32) _Float16 Ks[64][64];       // [key][d]
  __shared__ alignas(32) _Float16 VsT[64][64];      // [d][key]
  __shared__ alignas(32) _Float16 Ps[8][16][64];    // per-wave P strip [m][key]

  const int t = threadIdx.x;
  const int lane = t & 31, w = t >> 5;
  const int r = lane & 15, half = lane >> 4;
  const int q0 = blockIdx.x * 128;
  const int h = blockIdx.y;
  const size_t rowBase = (size_t)blockIdx.z * NSEQ;

  const size_t qkHead = (size_t)h * ROWS * HDIM;    // base of [h][row][d]
  const size_t vHead = (size_t)h * HDIM * ROWS;     // base of [h][d][row]

  // vectorized Q strip load: 1024 v8h chunks, 4 per thread, no guards
#pragma unroll
  for (int it = 0; it < 4; ++it) {
    const int v = t + it * 256;
    const int row = v >> 3, c8 = (v & 7) * 8;
    *(v8h*)&Qs[row][c8] =
        *(const v8h*)&q16[qkHead + (rowBase + q0 + row) * HDIM + c8];
  }

  float m_i[8], l_i[8], fs[8];
  v8f o[4];
#pragma unroll
  for (int i = 0; i < 8; ++i) { m_i[i] = -1e30f; l_i[i] = 0.f; }
#pragma unroll
  for (int dt = 0; dt < 4; ++dt) o[dt] = (v8f){0,0,0,0,0,0,0,0};

  __syncthreads();

  for (int kb = 0; kb < NSEQ / 64; ++kb) {
    // ---- stage K (natural) and V^T (already transposed) blocks, vectorized
#pragma unroll
    for (int it = 0; it < 2; ++it) {
      const int v = t + it * 256;            // [0,512)
      const int rr = v >> 3, c8 = (v & 7) * 8;
      *(v8h*)&Ks[rr][c8] =
          *(const v8h*)&k16[qkHead + (rowBase + kb * 64 + rr) * HDIM + c8];
      *(v8h*)&VsT[rr][c8] =
          *(const v8h*)&vT16[vHead + (size_t)rr * ROWS + rowBase + kb * 64 + c8];
    }
    __syncthreads();

    // ---- S = Q @ K^T (16x64 per wave, 4 tiles x 2 K-chunks) ----
    const _Float16* qp = &Qs[w * 16 + r][0];
    v16h qa0 = *(const v16h*)(qp + 16 * half);
    v16h qa1 = *(const v16h*)(qp + 32 + 16 * half);
    v8f st[4];
#pragma unroll
    for (int ct = 0; ct < 4; ++ct) {
      const _Float16* kp = &Ks[ct * 16 + r][0];
      v16h kb0 = *(const v16h*)(kp + 16 * half);
      v16h kb1 = *(const v16h*)(kp + 32 + 16 * half);
      v8f sacc = (v8f){0,0,0,0,0,0,0,0};
      sacc = wmma16x16x32(qa0, kb0, sacc);
      sacc = wmma16x16x32(qa1, kb1, sacc);
      st[ct] = sacc;
    }

    // ---- online softmax (row = vgpr i + 8*half; butterflies stay in-half)
#pragma unroll
    for (int ct = 0; ct < 4; ++ct)
#pragma unroll
      for (int i = 0; i < 8; ++i) st[ct][i] *= 0.125f;  // HD^-0.5

#pragma unroll
    for (int i = 0; i < 8; ++i) {
      float mx = fmaxf(fmaxf(st[0][i], st[1][i]), fmaxf(st[2][i], st[3][i]));
#pragma unroll
      for (int off = 1; off < 16; off <<= 1)
        mx = fmaxf(mx, __shfl_xor(mx, off, 32));
      const float mnew = fmaxf(m_i[i], mx);
      fs[i] = __expf(m_i[i] - mnew);
      m_i[i] = mnew;
      float rs = 0.f;
#pragma unroll
      for (int ct = 0; ct < 4; ++ct) {
        float p = __expf(st[ct][i] - mnew);
        st[ct][i] = p;
        rs += p;
      }
#pragma unroll
      for (int off = 1; off < 16; off <<= 1)
        rs += __shfl_xor(rs, off, 32);
      l_i[i] = l_i[i] * fs[i] + rs;
    }

#pragma unroll
    for (int dt = 0; dt < 4; ++dt)
#pragma unroll
      for (int i = 0; i < 8; ++i) o[dt][i] *= fs[i];

    // spill P (f16) to this wave's LDS strip in [m][key] order
#pragma unroll
    for (int ct = 0; ct < 4; ++ct)
#pragma unroll
      for (int i = 0; i < 8; ++i)
        Ps[w][i + 8 * half][ct * 16 + r] = (_Float16)st[ct][i];
    __syncthreads();

    // ---- O += P @ V ----
    const _Float16* pp = &Ps[w][r][0];
    v16h pa0 = *(const v16h*)(pp + 16 * half);
    v16h pa1 = *(const v16h*)(pp + 32 + 16 * half);
#pragma unroll
    for (int dt = 0; dt < 4; ++dt) {
      const _Float16* vp = &VsT[dt * 16 + r][0];
      v16h vb0 = *(const v16h*)(vp + 16 * half);
      v16h vb1 = *(const v16h*)(vp + 32 + 16 * half);
      o[dt] = wmma16x16x32(pa0, vb0, o[dt]);
      o[dt] = wmma16x16x32(pa1, vb1, o[dt]);
    }
    __syncthreads();  // protect Ks/VsT/Ps before next block's staging
  }

  // ---- normalize and write [row][h*64 + d] ----
#pragma unroll
  for (int i = 0; i < 8; ++i) {
    const float inv = 1.0f / l_i[i];
    const int row = q0 + w * 16 + i + 8 * half;
#pragma unroll
    for (int dt = 0; dt < 4; ++dt)
      attn16[(rowBase + row) * BDIM + h * HDIM + dt * 16 + r] =
          (_Float16)(o[dt][i] * inv);
  }
}

// ---------------------------------------------------------------------------
// Host-side orchestration
// ---------------------------------------------------------------------------
extern "C" void kernel_launch(void* const* d_in, const int* in_sizes, int n_in,
                              void* d_out, int out_size, void* d_ws, size_t ws_size,
                              hipStream_t stream) {
  const float* x      = (const float*)d_in[0];
  const float* ln1_g  = (const float*)d_in[1];
  const float* ln1_b  = (const float*)d_in[2];
  const float* qkv_w  = (const float*)d_in[3];
  const float* qkv_b  = (const float*)d_in[4];
  const float* proj_w = (const float*)d_in[5];
  const float* proj_b = (const float*)d_in[6];
  const float* ln2_g  = (const float*)d_in[7];
  const float* ln2_b  = (const float*)d_in[8];
  const float* ff1_w  = (const float*)d_in[9];
  const float* ff1_b  = (const float*)d_in[10];
  const float* ff2_w  = (const float*)d_in[11];
  const float* ff2_b  = (const float*)d_in[12];
  float* out = (float*)d_out;

  char* ws = (char*)d_ws;
  size_t off = 0;
  auto alloc = [&](size_t bytes) -> void* {
    void* p = ws + off;
    off = (off + bytes + 255) & ~(size_t)255;
    return p;
  };

  _Float16* wq16 = (_Float16*)alloc((size_t)BDIM * QKVCOLS * 2);
  _Float16* wp16 = (_Float16*)alloc((size_t)BDIM * BDIM * 2);
  _Float16* w116 = (_Float16*)alloc((size_t)BDIM * FFCOLS * 2);
  _Float16* w216 = (_Float16*)alloc((size_t)FFCOLS * BDIM * 2);
  _Float16* h16  = (_Float16*)alloc((size_t)ROWS * BDIM * 2);   // LN1/LN2 out
  _Float16* q16  = (_Float16*)alloc((size_t)NHEAD * ROWS * HDIM * 2);
  _Float16* k16  = (_Float16*)alloc((size_t)NHEAD * ROWS * HDIM * 2);
  _Float16* vT16 = (_Float16*)alloc((size_t)NHEAD * HDIM * ROWS * 2);
  _Float16* attn16 = (_Float16*)alloc((size_t)ROWS * BDIM * 2);
  _Float16* hff16  = (_Float16*)alloc((size_t)ROWS * FFCOLS * 2);
  float* x1 = (float*)alloc((size_t)ROWS * BDIM * 4);

  // 1) weight conversions fp32 -> fp16
  {
    int n;
    n = BDIM * QKVCOLS; cvt_f32_to_f16<<<(n + 255) / 256, 256, 0, stream>>>(qkv_w, wq16, n);
    n = BDIM * BDIM;    cvt_f32_to_f16<<<(n + 255) / 256, 256, 0, stream>>>(proj_w, wp16, n);
    n = BDIM * FFCOLS;  cvt_f32_to_f16<<<(n + 255) / 256, 256, 0, stream>>>(ff1_w, w116, n);
    n = FFCOLS * BDIM;  cvt_f32_to_f16<<<(n + 255) / 256, 256, 0, stream>>>(ff2_w, w216, n);
  }

  // 2) LN1
  ln_kernel<<<ROWS, 256, 0, stream>>>(x, ln1_g, ln1_b, h16);

  // 3) QKV GEMM -> deinterleaved per-head q/k/vT (f16)
  gemm_wmma<<<dim3(QKVCOLS / 128, ROWS / 128), 256, 0, stream>>>(
      h16, wq16, qkv_b, nullptr, nullptr, nullptr,
      q16, k16, vT16, ROWS, QKVCOLS, BDIM, 0, 2);

  // 4) Flash attention -> attn16
  attn_flash<<<dim3(NSEQ / 128, NHEAD, 2), 256, 0, stream>>>(q16, k16, vT16, attn16);

  // 5) proj GEMM + residual x -> x1 (f32)
  gemm_wmma<<<dim3(BDIM / 128, ROWS / 128), 256, 0, stream>>>(
      attn16, wp16, proj_b, x, x1, nullptr,
      nullptr, nullptr, nullptr, ROWS, BDIM, BDIM, 0, 0);

  // 6) LN2
  ln_kernel<<<ROWS, 256, 0, stream>>>(x1, ln2_g, ln2_b, h16);

  // 7) FF1 GEMM + exact GELU -> hff16 (f16)
  gemm_wmma<<<dim3(FFCOLS / 128, ROWS / 128), 256, 0, stream>>>(
      h16, w116, ff1_b, nullptr, nullptr, hff16,
      nullptr, nullptr, nullptr, ROWS, FFCOLS, BDIM, 1, 0);

  // 8) FF2 GEMM + residual x1 -> out (f32)
  gemm_wmma<<<dim3(BDIM / 128, ROWS / 128), 256, 0, stream>>>(
      hff16, w216, ff2_b, x1, out, nullptr,
      nullptr, nullptr, nullptr, ROWS, BDIM, FFCOLS, 0, 0);

  (void)in_sizes; (void)n_in; (void)out_size; (void)ws_size;
}
